// _RWKV6BlockFast_52450140619469
// MI455X (gfx1250) — matrix operand order repaired and checked by reading, not compiled, not run on hardware
//
#include <hip/hip_runtime.h>

// Problem constants (from reference): B=2, T=2048, D=1024, H=16, N=64
#define BB 2
#define TT 2048
#define DD 1024
#define HH 16
#define NN 64
#define MROWS (BB*TT)            // 4096 token rows

#define KTILE 32
#define MB    128
#define NB    64

typedef _Float16 v8h  __attribute__((ext_vector_type(8)));
typedef _Float16 v16h __attribute__((ext_vector_type(16)));
typedef float    v8f  __attribute__((ext_vector_type(8)));

__device__ __forceinline__ float sigmoidf_(float x) { return 1.0f / (1.0f + __expf(-x)); }

// Issue one per-lane 16-byte async global->LDS DMA (tracked by ASYNCcnt).
__device__ __forceinline__ void async_b128(const void* gptr, void* lptr) {
  asm volatile("global_load_async_to_lds_b128 %0, %1, off"
               :: "v"((unsigned int)(unsigned long long)lptr),
                  "v"((unsigned long long)gptr)
               : "memory");
}

// ---------------------------------------------------------------------------
// Weight conversion f32 -> f16 (grid-stride)
// ---------------------------------------------------------------------------
__global__ void f32_to_f16_kernel(const float* __restrict__ src,
                                  _Float16* __restrict__ dst, int n) {
  int idx = blockIdx.x * blockDim.x + threadIdx.x;
  int stride = gridDim.x * blockDim.x;
  for (int i = idx; i < n; i += stride) dst[i] = (_Float16)src[i];
}

// ---------------------------------------------------------------------------
// Fused LayerNorm + token-shift + 4-way mix, output f16 activations.
// One block (256 threads) per token row; recompute LN of previous row for xp.
// ---------------------------------------------------------------------------
__global__ __launch_bounds__(256)
void ln_mix_kernel(const float* __restrict__ x,
                   const float* __restrict__ ln_w, const float* __restrict__ ln_b,
                   const float* __restrict__ mk, const float* __restrict__ mv,
                   const float* __restrict__ mr, const float* __restrict__ mw,
                   _Float16* __restrict__ xk, _Float16* __restrict__ xv,
                   _Float16* __restrict__ xr, _Float16* __restrict__ xw) {
  const int m = blockIdx.x;          // 0..MROWS-1
  const int t = m & (TT - 1);
  const int tid = threadIdx.x;       // 0..255, 4 elems each

  const float4* row = (const float4*)(x + (size_t)m * DD);
  float4 c = row[tid];
  float4 p = make_float4(0.f, 0.f, 0.f, 0.f);
  if (t > 0) {
    const float4* prow = (const float4*)(x + (size_t)(m - 1) * DD);
    p = prow[tid];
  }

  float s0 = c.x + c.y + c.z + c.w;
  float q0 = c.x * c.x + c.y * c.y + c.z * c.z + c.w * c.w;
  float s1 = p.x + p.y + p.z + p.w;
  float q1 = p.x * p.x + p.y * p.y + p.z * p.z + p.w * p.w;

  // wave32 reductions
  #pragma unroll
  for (int off = 16; off > 0; off >>= 1) {
    s0 += __shfl_down(s0, off, 32);
    q0 += __shfl_down(q0, off, 32);
    s1 += __shfl_down(s1, off, 32);
    q1 += __shfl_down(q1, off, 32);
  }
  __shared__ float red[4][8];
  __shared__ float stats[4];
  const int wave = tid >> 5, lane = tid & 31;
  if (lane == 0) { red[0][wave] = s0; red[1][wave] = q0; red[2][wave] = s1; red[3][wave] = q1; }
  __syncthreads();
  if (tid == 0) {
    float a = 0.f, b = 0.f, cc = 0.f, dd = 0.f;
    #pragma unroll
    for (int w = 0; w < 8; w++) { a += red[0][w]; b += red[1][w]; cc += red[2][w]; dd += red[3][w]; }
    float mu0 = a * (1.0f / DD), var0 = b * (1.0f / DD) - mu0 * mu0;
    float mu1 = cc * (1.0f / DD), var1 = dd * (1.0f / DD) - mu1 * mu1;
    stats[0] = mu0; stats[1] = rsqrtf(var0 + 1e-5f);
    stats[2] = mu1; stats[3] = rsqrtf(var1 + 1e-5f);
  }
  __syncthreads();
  const float mu0 = stats[0], rs0 = stats[1], mu1 = stats[2], rs1 = stats[3];

  float cc4[4] = { c.x, c.y, c.z, c.w };
  float pp4[4] = { p.x, p.y, p.z, p.w };
  const int dbase = tid * 4;
  #pragma unroll
  for (int j = 0; j < 4; j++) {
    const int d = dbase + j;
    const float lw = ln_w[d], lb = ln_b[d];
    const float xn = (cc4[j] - mu0) * rs0 * lw + lb;
    const float xp = (t > 0) ? ((pp4[j] - mu1) * rs1 * lw + lb) : 0.0f;
    const size_t o = (size_t)m * DD + d;
    const float dx = xn - xp;
    xk[o] = (_Float16)(xp + sigmoidf_(mk[d]) * dx);
    xv[o] = (_Float16)(xp + sigmoidf_(mv[d]) * dx);
    xr[o] = (_Float16)(xp + sigmoidf_(mr[d]) * dx);
    xw[o] = (_Float16)(xp + sigmoidf_(mw[d]) * dx);
  }
}

// ---------------------------------------------------------------------------
// WMMA GEMM: C[M,1024] = act( A16[M,1024] @ W16[1024,1024]^T )
// Block: 256 threads = 8 waves; block tile 128(M) x 64(N); wave tile 32x32
// (2x2 v_wmma_f32_16x16x32_f16). Tiles staged global->LDS with async DMA
// (GLOBAL_LOAD_ASYNC_TO_LDS_B128, ASYNCcnt), double-buffered; fragments read
// from LDS (ds_load_b128). Each thread issues exactly 3 async b128 per tile.
// mode: 0 = identity, 1 = sigmoid, 2 = decay: exp(-exp(clip(bias[n]+v,-10,10)))
// ---------------------------------------------------------------------------
__global__ __launch_bounds__(256)
void gemm_wmma_kernel(const _Float16* __restrict__ A,
                      const _Float16* __restrict__ W,
                      float* __restrict__ C,
                      const float* __restrict__ bias,
                      int mode) {
  const int K = DD, Nc = DD;
  __shared__ _Float16 ldsA[2][MB * KTILE];   // 2 x 8KB
  __shared__ _Float16 ldsB[2][NB * KTILE];   // 2 x 4KB

  const int tid  = threadIdx.x;
  const int lane = tid & 31;
  const int wave = tid >> 5;           // 0..7
  const int wm = wave & 3, wn = wave >> 2;
  const int blockN = blockIdx.x * NB;
  const int blockM = blockIdx.y * MB;
  const int half = lane >> 4;          // 0 or 1 (K-half select)
  const int l16  = lane & 15;

  // copy-lane decomposition: thread -> (row, 8-half column chunk)
  const int crow = tid >> 2;           // 0..63
  const int ccol = (tid & 3) * 8;      // 0,8,16,24

  const _Float16* Abase = A + (size_t)(blockM + crow) * K + ccol;
  const _Float16* Wbase = W + (size_t)(blockN + crow) * K + ccol;

  auto issue_tile = [&](int buf, int k0) {
    // A tile: 128x32 halves, two 64-row chunks per thread
    async_b128(Abase + k0,                 &ldsA[buf][crow * KTILE + ccol]);
    async_b128(Abase + (size_t)64 * K + k0, &ldsA[buf][(crow + 64) * KTILE + ccol]);
    // B tile: 64x32 halves, one chunk per thread
    async_b128(Wbase + k0,                 &ldsB[buf][crow * KTILE + ccol]);
  };

  v8f acc[2][2] = {};

  issue_tile(0, 0);
  const int KT = K / KTILE;            // 32
  for (int kt = 0; kt < KT; kt++) {
    const int cur = kt & 1;
    if (kt + 1 < KT) {
      issue_tile(cur ^ 1, (kt + 1) * KTILE);
      asm volatile("s_wait_asynccnt 0x3" ::: "memory");   // current tile landed
    } else {
      asm volatile("s_wait_asynccnt 0x0" ::: "memory");
    }
    __syncthreads();                   // all waves' DMA visible

    v16h a[2], b[2];
    #pragma unroll
    for (int i = 0; i < 2; i++) {
      // A fragment rows wm*32 + i*16 + l16; K runs half*8 and half*8+16
      const int r = (wm * 32 + i * 16 + l16) * KTILE + half * 8;
      v8h lo = *(const v8h*)&ldsA[cur][r];
      v8h hi = *(const v8h*)&ldsA[cur][r + 16];
      #pragma unroll
      for (int e = 0; e < 8; e++) { a[i][e] = lo[e]; a[i][8 + e] = hi[e]; }
    }
    #pragma unroll
    for (int j = 0; j < 2; j++) {
      // B fragment: col group wn*32 + j*16 + l16; K runs half*8 and half*8+16
      const int r = (wn * 32 + j * 16 + l16) * KTILE + half * 8;
      v8h lo = *(const v8h*)&ldsB[cur][r];
      v8h hi = *(const v8h*)&ldsB[cur][r + 16];
      #pragma unroll
      for (int e = 0; e < 8; e++) { b[j][e] = lo[e]; b[j][8 + e] = hi[e]; }
    }
    #pragma unroll
    for (int i = 0; i < 2; i++)
      #pragma unroll
      for (int j = 0; j < 2; j++)
        acc[i][j] = __builtin_amdgcn_wmma_f32_16x16x32_f16(
            false, a[i], false, b[j], (short)0, acc[i][j], false, false);

    __syncthreads();                   // done reading before buffer reuse
  }

  // Epilogue: C/D layout — VGPR rr holds row rr (lanes 0-15) / 8+rr (16-31).
  #pragma unroll
  for (int i = 0; i < 2; i++) {
    const int mbase = blockM + wm * 32 + i * 16 + half * 8;
    #pragma unroll
    for (int j = 0; j < 2; j++) {
      const int n = blockN + wn * 32 + j * 16 + l16;
      const float bval = (mode == 2) ? bias[n] : 0.0f;
      #pragma unroll
      for (int rr = 0; rr < 8; rr++) {
        float vv = acc[i][j][rr];
        if (mode == 1) {
          vv = sigmoidf_(vv);
        } else if (mode == 2) {
          float wv = bval + vv;
          wv = fminf(10.0f, fmaxf(-10.0f, wv));
          vv = expf(-expf(wv));
        }
        C[(size_t)(mbase + rr) * Nc + n] = vv;
      }
    }
  }
}

// ---------------------------------------------------------------------------
// RWKV6 sequential scan. One block per (b,h); 256 threads hold the 64x64
// state in registers (thread (i, jg) owns S[jg*16..+16][i]).
// y_i = sum_j r_j * (S[j,i] + u_j*k_j*v_i);  S[j,i] = d_j*S[j,i] + k_j*v_i
// ---------------------------------------------------------------------------
__global__ __launch_bounds__(256)
void rwkv_scan_kernel(const float* __restrict__ r, const float* __restrict__ k,
                      const float* __restrict__ v, const float* __restrict__ dec,
                      const float* __restrict__ u, float* __restrict__ y) {
  const int bh = blockIdx.x;           // 0..31
  const int b = bh >> 4, h = bh & 15;
  const int tid = threadIdx.x;
  const int i  = tid & 63;             // output column
  const int jg = tid >> 6;             // 0..3: 16-row group

  __shared__ float rs[64], ks[64], ds[64];
  __shared__ float part[256];

  float S[16];
  #pragma unroll
  for (int q = 0; q < 16; q++) S[q] = 0.0f;
  float uj[16];
  #pragma unroll
  for (int q = 0; q < 16; q++) uj[q] = u[h * 64 + jg * 16 + q];

  size_t off = (size_t)b * TT * DD + (size_t)h * 64;
  for (int t = 0; t < TT; t++) {
    if (tid < 64) {
      rs[tid] = r[off + tid];
      ks[tid] = k[off + tid];
      ds[tid] = dec[off + tid];
    }
    __builtin_prefetch(&r[off + DD + i], 0, 3);   // near-scope prefetch of t+1
    __builtin_prefetch(&v[off + DD + i], 0, 3);
    __syncthreads();
    const float vi = v[off + i];
    float acc = 0.0f;
    #pragma unroll
    for (int q = 0; q < 16; q++) {
      const int j = jg * 16 + q;
      const float kv = ks[j] * vi;
      acc = fmaf(rs[j], S[q] + uj[q] * kv, acc);
      S[q] = fmaf(ds[j], S[q], kv);
    }
    part[tid] = acc;
    __syncthreads();
    if (jg == 0) y[off + i] = part[i] + part[i + 64] + part[i + 128] + part[i + 192];
    __syncthreads();
    off += DD;
  }
}

// ---------------------------------------------------------------------------
// gy = (f16)(g * y), elementwise
// ---------------------------------------------------------------------------
__global__ void gy_kernel(const float* __restrict__ g, const float* __restrict__ y,
                          _Float16* __restrict__ gy, int n) {
  int idx = blockIdx.x * blockDim.x + threadIdx.x;
  int stride = gridDim.x * blockDim.x;
  for (int i = idx; i < n; i += stride) gy[i] = (_Float16)(g[i] * y[i]);
}

// ---------------------------------------------------------------------------
// Host launch
// ---------------------------------------------------------------------------
extern "C" void kernel_launch(void* const* d_in, const int* in_sizes, int n_in,
                              void* d_out, int out_size, void* d_ws, size_t ws_size,
                              hipStream_t stream) {
  const float* x   = (const float*)d_in[0];
  const float* lnw = (const float*)d_in[1];
  const float* lnb = (const float*)d_in[2];
  const float* mk  = (const float*)d_in[3];
  const float* mv  = (const float*)d_in[4];
  const float* mr  = (const float*)d_in[5];
  const float* mw  = (const float*)d_in[6];
  const float* Wr  = (const float*)d_in[7];
  const float* Wk  = (const float*)d_in[8];
  const float* Wv  = (const float*)d_in[9];
  const float* Ww  = (const float*)d_in[10];
  const float* Wg  = (const float*)d_in[11];
  const float* Wo  = (const float*)d_in[12];
  const float* td  = (const float*)d_in[13];  // time_decay [D]
  const float* tf  = (const float*)d_in[14];  // time_first [H,N]
  float* out = (float*)d_out;

  const size_t MD = (size_t)MROWS * DD;      // 4M elems
  const size_t WW = (size_t)DD * DD;         // 1M elems

  char* ws = (char*)d_ws;
  size_t off = 0;
  auto carve = [&](size_t bytes) -> void* {
    void* p = ws + off;
    off += (bytes + 255) & ~(size_t)255;
    return p;
  };

  _Float16* xk16 = (_Float16*)carve(MD * 2);
  _Float16* xv16 = (_Float16*)carve(MD * 2);
  _Float16* xr16 = (_Float16*)carve(MD * 2);
  _Float16* xw16 = (_Float16*)carve(MD * 2);
  _Float16* Wr16 = (_Float16*)carve(WW * 2);
  _Float16* Wk16 = (_Float16*)carve(WW * 2);
  _Float16* Wv16 = (_Float16*)carve(WW * 2);
  _Float16* Ww16 = (_Float16*)carve(WW * 2);
  _Float16* Wg16 = (_Float16*)carve(WW * 2);
  _Float16* Wo16 = (_Float16*)carve(WW * 2);
  float* r32 = (float*)carve(MD * 4);
  float* k32 = (float*)carve(MD * 4);
  float* v32 = (float*)carve(MD * 4);
  float* d32 = (float*)carve(MD * 4);
  float* g32 = (float*)carve(MD * 4);
  float* y32 = (float*)carve(MD * 4);
  _Float16* gy16 = (_Float16*)carve(MD * 2);
  (void)ws_size; (void)n_in; (void)in_sizes; (void)out_size;

  // 1) weights -> f16
  {
    dim3 g(512), bsz(256);
    f32_to_f16_kernel<<<g, bsz, 0, stream>>>(Wr, Wr16, (int)WW);
    f32_to_f16_kernel<<<g, bsz, 0, stream>>>(Wk, Wk16, (int)WW);
    f32_to_f16_kernel<<<g, bsz, 0, stream>>>(Wv, Wv16, (int)WW);
    f32_to_f16_kernel<<<g, bsz, 0, stream>>>(Ww, Ww16, (int)WW);
    f32_to_f16_kernel<<<g, bsz, 0, stream>>>(Wg, Wg16, (int)WW);
    f32_to_f16_kernel<<<g, bsz, 0, stream>>>(Wo, Wo16, (int)WW);
  }

  // 2) fused LN + token-shift + mix
  ln_mix_kernel<<<dim3(MROWS), dim3(256), 0, stream>>>(
      x, lnw, lnb, mk, mv, mr, mw, xk16, xv16, xr16, xw16);

  // 3) five projection GEMMs (fused activations)
  dim3 ggrid(DD / NB, MROWS / MB), gblk(256);
  gemm_wmma_kernel<<<ggrid, gblk, 0, stream>>>(xr16, Wr16, r32, nullptr, 1); // r = sigmoid
  gemm_wmma_kernel<<<ggrid, gblk, 0, stream>>>(xk16, Wk16, k32, nullptr, 0); // k
  gemm_wmma_kernel<<<ggrid, gblk, 0, stream>>>(xv16, Wv16, v32, nullptr, 0); // v
  gemm_wmma_kernel<<<ggrid, gblk, 0, stream>>>(xw16, Ww16, d32, td,      2); // decay
  gemm_wmma_kernel<<<ggrid, gblk, 0, stream>>>(xr16, Wg16, g32, nullptr, 1); // g = sigmoid

  // 4) sequential RWKV scan over T
  rwkv_scan_kernel<<<dim3(BB * HH), dim3(256), 0, stream>>>(r32, k32, v32, d32, tf, y32);

  // 5) gy = g * y (f16)
  gy_kernel<<<dim3(4096), dim3(256), 0, stream>>>(g32, y32, gy16, (int)MD);

  // 6) output projection into d_out
  gemm_wmma_kernel<<<ggrid, gblk, 0, stream>>>(gy16, Wo16, out, nullptr, 0);
}
